// RopeAlbertAttention_31086973289073
// MI455X (gfx1250) — matrix operand order repaired and checked by reading, not compiled, not run on hardware
//
#include <hip/hip_runtime.h>
#include <hip/hip_bf16.h>
#include <stdint.h>

#define B_    2
#define SEQ   2048
#define HID_  1024
#define NH    16
#define DH    64
#define MROWS (B_*SEQ)   // 4096 token rows

typedef __attribute__((ext_vector_type(16))) __bf16 v16bf;
typedef __attribute__((ext_vector_type(8)))  float  v8f;

union FragU { v16bf v; uint4 q[2]; };

__device__ __forceinline__ unsigned short f32_to_bf16(float f) {
  unsigned int u = __float_as_uint(f);
  unsigned int r = 0x7FFFu + ((u >> 16) & 1u);   // round-to-nearest-even
  return (unsigned short)((u + r) >> 16);
}

// ---- WMMA fragment loaders (CDNA5 ISA 7.12.2 layouts, wave32) ----
// A 16x32 bf16: lane l(0..15)=row M; half0 holds K {kb+0..7, kb+16..23},
//               half1 holds K {kb+8..15, kb+24..31}
__device__ __forceinline__ void load_frag_a(FragU& f, const unsigned short* base,
                                            int rowbase, int ld, int kb, int lane) {
  int hf = lane >> 4, l = lane & 15;
  const unsigned short* p = base + (size_t)(rowbase + l) * (size_t)ld + kb + hf * 8;
  f.q[0] = *(const uint4*)(p);
  f.q[1] = *(const uint4*)(p + 16);
}

// B 32x16 bf16: lane l(0..15)=col N; half0 holds K {kb+0..7, kb+16..23},
//               half1 holds K {kb+8..15, kb+24..31}.
// Source is row-major [N, K] (i.e. rows of W give W^T operand; rows of K-matrix
// give K^T operand; rows of V^T give V operand).
__device__ __forceinline__ void load_frag_b(FragU& f, const unsigned short* base,
                                            int nbase, int ld, int kb, int lane) {
  int hf = lane >> 4, l = lane & 15;
  const unsigned short* p = base + (size_t)(nbase + l) * (size_t)ld + kb + hf * 8;
  f.q[0] = *(const uint4*)(p);
  f.q[1] = *(const uint4*)(p + 16);
}

__device__ __forceinline__ v8f wmma_bf16(const FragU& a, const FragU& b, v8f c) {
  return __builtin_amdgcn_wmma_f32_16x16x32_bf16(false, a.v, false, b.v,
                                                 (short)0, c, false, false);
}

// --------------------------------------------------------------------------
// Kernel 0: fp32 -> bf16 conversion
// --------------------------------------------------------------------------
__global__ void __launch_bounds__(256) cvt_bf16(const float* __restrict__ in,
                                                unsigned short* __restrict__ out, int n) {
  int i = blockIdx.x * 256 + threadIdx.x;
  if (i < n) out[i] = f32_to_bf16(in[i]);
}

// --------------------------------------------------------------------------
// Kernel 1: projection GEMM  C[4096,1024] = A[4096,1024] @ W[1024,1024]^T + b
//   mode 0: store bf16 transposed-per-head  [B,H,64,S]     (V path)
//   mode 1: RoPE, store bf16 per-head       [B,H,S,64]     (Q/K path)
//   mode 2: + residual, store fp32          [4096,1024]    (out-proj path)
// Block 256 thr (8 waves), tile 128x64; wave tile 32x32 (2x2 WMMA).
// Operands fetched straight from global: whole working set is L2-resident.
// --------------------------------------------------------------------------
__global__ void __launch_bounds__(256) proj_gemm(
    const unsigned short* __restrict__ A,     // bf16 [4096,1024]
    const unsigned short* __restrict__ W,     // bf16 [1024,1024] row-major (out-ch major)
    const float* __restrict__ bias,           // [1024]
    const float* __restrict__ sinus,          // [S,64] sin||cos (mode 1) or null
    const float* __restrict__ resid,          // fp32 [4096,1024] (mode 2) or null
    unsigned short* __restrict__ out_bf,      // bf16 dst (mode 0/1)
    float* __restrict__ out_f,                // fp32 dst (mode 2)
    int mode)
{
  int lane = threadIdx.x & 31;
  int wave = threadIdx.x >> 5;
  int wr = wave & 3, wc = wave >> 2;
  int rbase = blockIdx.y * 128 + wr * 32;
  int cbase = blockIdx.x * 64  + wc * 32;

  v8f acc[2][2];
#pragma unroll
  for (int i = 0; i < 2; i++)
#pragma unroll
    for (int j = 0; j < 2; j++)
#pragma unroll
      for (int e = 0; e < 8; e++) acc[i][j][e] = 0.0f;

  for (int kb = 0; kb < HID_; kb += 32) {
    FragU a0, a1, b0, b1;
    load_frag_a(a0, A, rbase,      HID_, kb, lane);
    load_frag_a(a1, A, rbase + 16, HID_, kb, lane);
    load_frag_b(b0, W, cbase,      HID_, kb, lane);
    load_frag_b(b1, W, cbase + 16, HID_, kb, lane);
    acc[0][0] = wmma_bf16(a0, b0, acc[0][0]);
    acc[0][1] = wmma_bf16(a0, b1, acc[0][1]);
    acc[1][0] = wmma_bf16(a1, b0, acc[1][0]);
    acc[1][1] = wmma_bf16(a1, b1, acc[1][1]);
  }

  int hf = lane >> 4, l = lane & 15;
#pragma unroll
  for (int i = 0; i < 2; i++) {
#pragma unroll
    for (int j = 0; j < 2; j++) {
      int c = cbase + j * 16 + l;          // hidden column
      float bv = bias[c];
      int h = c >> 6, d = c & 63;          // head, dim-in-head
#pragma unroll
      for (int jj = 0; jj < 8; jj++) {
        int r = rbase + i * 16 + hf * 8 + jj;   // token row
        int bidx = r >> 11;                     // / SEQ
        int s = r & (SEQ - 1);
        float val = acc[i][j][jj] + bv;
        if (mode == 2) {
          val += resid[(size_t)r * HID_ + c];
          out_f[(size_t)r * HID_ + c] = val;
        } else if (mode == 1) {
          // interleaved RoPE: pair partner sits in adjacent lane of C layout
          float partner = __shfl_xor(val, 1, 32);
          float sn = sinus[(size_t)s * DH + (d >> 1)];
          float cs = sinus[(size_t)s * DH + 32 + (d >> 1)];
          float rot = (d & 1) ? partner : -partner;
          float v2 = val * cs + rot * sn;
          out_bf[(((size_t)bidx * NH + h) * SEQ + s) * DH + d] = f32_to_bf16(v2);
        } else {
          // V: store transposed per head -> [B,H,64,S] so PV B-operand is row loads
          out_bf[(((size_t)bidx * NH + h) * DH + d) * SEQ + s] = f32_to_bf16(val);
        }
      }
    }
  }
}

// --------------------------------------------------------------------------
// Kernel 2: flash attention over one (b,h), 64 Q-rows per block.
// 4 waves x 16 Q-rows; K/V tiles (64 keys) staged in LDS; online softmax.
// --------------------------------------------------------------------------
__global__ void __launch_bounds__(128) attn(
    const unsigned short* __restrict__ qbf,   // bf16 [B,H,S,64]  (RoPE'd)
    const unsigned short* __restrict__ kbf,   // bf16 [B,H,S,64]  (RoPE'd)
    const unsigned short* __restrict__ vtbf,  // bf16 [B,H,64,S]  (transposed)
    const float* __restrict__ maskp,          // [B,S] additive
    unsigned short* __restrict__ ctx)         // bf16 [4096,1024]
{
  __shared__ unsigned short Ks[64 * 64];      // 8 KB
  __shared__ unsigned short Vs[64 * 64];      // 8 KB  (rows = d, cols = key)
  __shared__ unsigned short Ps[4][16 * 64];   // 8 KB  per-wave P scratch

  int qt = blockIdx.x & 31;                   // SQ/64 tiles
  int bh = blockIdx.x >> 5;
  int b  = bh >> 4, h = bh & 15;

  int lane = threadIdx.x & 31;
  int wave = threadIdx.x >> 5;
  int hf = lane >> 4, l = lane & 15;

  const unsigned short* qb = qbf  + (size_t)bh * SEQ * DH;
  const unsigned short* kb = kbf  + (size_t)bh * SEQ * DH;
  const unsigned short* vt = vtbf + (size_t)bh * DH * SEQ;

  int qbase = qt * 64 + wave * 16;

  FragU aq[2];
  load_frag_a(aq[0], qb, qbase, DH, 0,  lane);
  load_frag_a(aq[1], qb, qbase, DH, 32, lane);

  v8f o[4];
#pragma unroll
  for (int t = 0; t < 4; t++)
#pragma unroll
    for (int e = 0; e < 8; e++) o[t][e] = 0.0f;
  float mrow[8], lrow[8];
#pragma unroll
  for (int jj = 0; jj < 8; jj++) { mrow[jj] = -1e30f; lrow[jj] = 0.0f; }

  const float scale = 0.125f;  // 1/sqrt(64)

  for (int kt = 0; kt < SEQ; kt += 64) {
    // ---- stage K (row-major keys) and V^T (row-major d) tiles into LDS ----
    {
      int rr = threadIdx.x >> 1, part = threadIdx.x & 1;   // 64 rows x 2 halves
      const uint4* sk = (const uint4*)(kb + (size_t)(kt + rr) * DH + part * 32);
      uint4* dk = (uint4*)(Ks + rr * 64 + part * 32);
      dk[0] = sk[0]; dk[1] = sk[1]; dk[2] = sk[2]; dk[3] = sk[3];
      const uint4* sv = (const uint4*)(vt + (size_t)rr * SEQ + kt + part * 32);
      uint4* dv = (uint4*)(Vs + rr * 64 + part * 32);
      dv[0] = sv[0]; dv[1] = sv[1]; dv[2] = sv[2]; dv[3] = sv[3];
    }
    __syncthreads();

    // ---- S = Q K^T (16q x 64k) ----
    v8f sc[4];
#pragma unroll
    for (int n = 0; n < 4; n++)
#pragma unroll
      for (int e = 0; e < 8; e++) sc[n][e] = 0.0f;
#pragma unroll
    for (int kc = 0; kc < 2; kc++) {
#pragma unroll
      for (int n = 0; n < 4; n++) {
        FragU fb;
        load_frag_b(fb, Ks, n * 16, 64, kc * 32, lane);
        sc[n] = wmma_bf16(aq[kc], fb, sc[n]);
      }
    }

    // ---- scale + mask + online softmax (rows live per (half,jj)) ----
    float mval[4];
#pragma unroll
    for (int n = 0; n < 4; n++) mval[n] = maskp[(size_t)b * SEQ + kt + n * 16 + l];
#pragma unroll
    for (int jj = 0; jj < 8; jj++) {
      float mx = -1e30f;
#pragma unroll
      for (int n = 0; n < 4; n++) {
        float t = sc[n][jj] * scale + mval[n];
        sc[n][jj] = t;
        mx = fmaxf(mx, t);
      }
#pragma unroll
      for (int off = 1; off < 16; off <<= 1) mx = fmaxf(mx, __shfl_xor(mx, off, 32));
      float mnew  = fmaxf(mrow[jj], mx);
      float alpha = __expf(mrow[jj] - mnew);
      float rs = 0.0f;
#pragma unroll
      for (int n = 0; n < 4; n++) {
        float p = __expf(sc[n][jj] - mnew);
        sc[n][jj] = p;
        rs += p;
      }
#pragma unroll
      for (int off = 1; off < 16; off <<= 1) rs += __shfl_xor(rs, off, 32);
      lrow[jj] = lrow[jj] * alpha + rs;
      mrow[jj] = mnew;
#pragma unroll
      for (int t = 0; t < 4; t++) o[t][jj] *= alpha;
    }

    // ---- reshape P: C layout -> A layout via per-wave LDS scratch ----
#pragma unroll
    for (int n = 0; n < 4; n++)
#pragma unroll
      for (int jj = 0; jj < 8; jj++)
        Ps[wave][(hf * 8 + jj) * 64 + n * 16 + l] = f32_to_bf16(sc[n][jj]);
    __syncthreads();

    FragU ap[2];
    load_frag_a(ap[0], Ps[wave], 0, 64, 0,  lane);
    load_frag_a(ap[1], Ps[wave], 0, 64, 32, lane);

    // ---- O += P V (V^T tile rows are the B-operand "n" rows) ----
#pragma unroll
    for (int t = 0; t < 4; t++) {
#pragma unroll
      for (int kc = 0; kc < 2; kc++) {
        FragU fv;
        load_frag_b(fv, Vs, t * 16, 64, kc * 32, lane);
        o[t] = wmma_bf16(ap[kc], fv, o[t]);
      }
    }
    __syncthreads();
  }

  // ---- normalize rows and scatter to token-major bf16 ctx [4096,1024] ----
#pragma unroll
  for (int jj = 0; jj < 8; jj++) {
    float inv = 1.0f / lrow[jj];
    int qrow = qbase + hf * 8 + jj;
#pragma unroll
    for (int t = 0; t < 4; t++) {
      float val = o[t][jj] * inv;
      ctx[((size_t)b * SEQ + qrow) * HID_ + h * DH + t * 16 + l] = f32_to_bf16(val);
    }
  }
}

// --------------------------------------------------------------------------
// Kernel 3: LayerNorm over rows of 1024 (one wave per row)
// --------------------------------------------------------------------------
__global__ void __launch_bounds__(256) layernorm(
    const float* __restrict__ x, const float* __restrict__ gamma,
    const float* __restrict__ beta, float* __restrict__ out)
{
  int wave = threadIdx.x >> 5, lane = threadIdx.x & 31;
  int row = blockIdx.x * 8 + wave;
  const float* xr = x + (size_t)row * HID_;
  float v[32];
  float sum = 0.0f;
#pragma unroll
  for (int i = 0; i < 32; i++) { v[i] = xr[lane + i * 32]; sum += v[i]; }
#pragma unroll
  for (int off = 1; off < 32; off <<= 1) sum += __shfl_xor(sum, off, 32);
  float mu = sum * (1.0f / HID_);
  float var = 0.0f;
#pragma unroll
  for (int i = 0; i < 32; i++) { float d = v[i] - mu; var += d * d; }
#pragma unroll
  for (int off = 1; off < 32; off <<= 1) var += __shfl_xor(var, off, 32);
  var *= (1.0f / HID_);
  float inv = rsqrtf(var + 1e-12f);
#pragma unroll
  for (int i = 0; i < 32; i++) {
    int c = lane + i * 32;
    out[(size_t)row * HID_ + c] = (v[i] - mu) * inv * gamma[c] + beta[c];
  }
}

// --------------------------------------------------------------------------
extern "C" void kernel_launch(void* const* d_in, const int* in_sizes, int n_in,
                              void* d_out, int out_size, void* d_ws, size_t ws_size,
                              hipStream_t stream) {
  const float* hidden = (const float*)d_in[0];
  const float* enc    = (const float*)d_in[1];
  const float* maskp  = (const float*)d_in[2];
  const float* sinq   = (const float*)d_in[3];
  const float* sink   = (const float*)d_in[4];
  const float* Wq = (const float*)d_in[5];  const float* bq = (const float*)d_in[6];
  const float* Wk = (const float*)d_in[7];  const float* bk = (const float*)d_in[8];
  const float* Wv = (const float*)d_in[9];  const float* bv = (const float*)d_in[10];
  const float* Wd = (const float*)d_in[11]; const float* bd = (const float*)d_in[12];
  const float* gamma = (const float*)d_in[13];
  const float* beta  = (const float*)d_in[14];

  const size_t MT = (size_t)MROWS * HID_;   // 4M elements
  const size_t WT = (size_t)HID_ * HID_;    // 1M elements

  char* p = (char*)d_ws;
  unsigned short* hbf   = (unsigned short*)p; p += MT * 2;
  unsigned short* ebf   = (unsigned short*)p; p += MT * 2;
  unsigned short* wqbf  = (unsigned short*)p; p += WT * 2;
  unsigned short* wkbf  = (unsigned short*)p; p += WT * 2;
  unsigned short* wvbf  = (unsigned short*)p; p += WT * 2;
  unsigned short* wdbf  = (unsigned short*)p; p += WT * 2;
  unsigned short* qbf   = (unsigned short*)p; p += MT * 2;
  unsigned short* kbfb  = (unsigned short*)p; p += MT * 2;
  unsigned short* vtbf  = (unsigned short*)p; p += MT * 2;
  unsigned short* ctxbf = (unsigned short*)p; p += MT * 2;
  float*          xres  = (float*)p;          p += MT * 4;

  // fp32 -> bf16 operand conversion
  cvt_bf16<<<(int)((MT + 255) / 256), 256, 0, stream>>>(hidden, hbf, (int)MT);
  cvt_bf16<<<(int)((MT + 255) / 256), 256, 0, stream>>>(enc,    ebf, (int)MT);
  cvt_bf16<<<(int)((WT + 255) / 256), 256, 0, stream>>>(Wq, wqbf, (int)WT);
  cvt_bf16<<<(int)((WT + 255) / 256), 256, 0, stream>>>(Wk, wkbf, (int)WT);
  cvt_bf16<<<(int)((WT + 255) / 256), 256, 0, stream>>>(Wv, wvbf, (int)WT);
  cvt_bf16<<<(int)((WT + 255) / 256), 256, 0, stream>>>(Wd, wdbf, (int)WT);

  dim3 grid(HID_ / 64, MROWS / 128);   // (16, 32)
  // Q = RoPE(hidden @ Wq^T + bq) -> [B,H,S,64]
  proj_gemm<<<grid, 256, 0, stream>>>(hbf, wqbf, bq, sinq, nullptr, qbf,  nullptr, 1);
  // K = RoPE(enc @ Wk^T + bk)    -> [B,H,S,64]
  proj_gemm<<<grid, 256, 0, stream>>>(ebf, wkbf, bk, sink, nullptr, kbfb, nullptr, 1);
  // V = enc @ Wv^T + bv          -> [B,H,64,S] (transposed)
  proj_gemm<<<grid, 256, 0, stream>>>(ebf, wvbf, bv, nullptr, nullptr, vtbf, nullptr, 0);

  // attention -> ctx bf16 [4096,1024]
  attn<<<B_ * NH * (SEQ / 64), 128, 0, stream>>>(qbf, kbfb, vtbf, maskp, ctxbf);

  // out-proj + residual -> xres fp32
  proj_gemm<<<grid, 256, 0, stream>>>(ctxbf, wdbf, bd, nullptr, hidden, nullptr, xres, 2);

  // LayerNorm -> d_out fp32
  layernorm<<<MROWS / 8, 256, 0, stream>>>(xres, gamma, beta, (float*)d_out);
}
